// graph_encode_7035156431104
// MI455X (gfx1250) — matrix operand order repaired and checked by reading, not compiled, not run on hardware
//
#include <hip/hip_runtime.h>
#include <hip/hip_bf16.h>

// ---------------------------------------------------------------------------
// CDNA5 (gfx1250) fused graph-transformer encoder.
// B=16, E=400, R=624, N=1024, HSZ=512, HEADS=4 (dk=128), PROP=2, DFF=2048.
// f16 activations -> v_wmma_f32_16x16x32_f16 (f32 accum). Bandwidth-bound on
// MI455X (23.3 TB/s): scores never leave LDS, bias/PReLU/residual fused,
// LDS double-buffered with one barrier per K-step, global_prefetch_b8 on the
// streaming operands.
// ---------------------------------------------------------------------------

typedef _Float16 v16h __attribute__((ext_vector_type(16)));
typedef _Float16 v8h  __attribute__((ext_vector_type(8)));
typedef float    v8f  __attribute__((ext_vector_type(8)));
typedef float    v4f  __attribute__((ext_vector_type(4)));

#define Bn   16
#define En   400
#define Rn   624
#define Nn   1024
#define HSZn 512
#define DKn  128
#define DFFn 2048

// ------------------------------- gather ------------------------------------
__global__ void gather_kernel(const float* __restrict__ ents,
                              const int*   __restrict__ rels,
                              const float* __restrict__ renc,
                              float* __restrict__ Xf,
                              _Float16* __restrict__ Xh) {
  int bn = blockIdx.x;              // b*1024 + n
  int b = bn >> 10;
  int n = bn & 1023;
  const float* src;
  if (n < En) {
    src = ents + ((size_t)b * En + n) * HSZn;
  } else {
    int r = rels[b * Rn + (n - En)];
    src = renc + (size_t)r * HSZn;
  }
  size_t dst = (size_t)bn * HSZn;
  for (int c = threadIdx.x; c < HSZn; c += 256) {
    float v = src[c];
    Xf[dst + c] = v;
    Xh[dst + c] = (_Float16)v;
  }
}

// ------------------------------ generic GEMM -------------------------------
// C[M,Nc] = A[M,K](f16) @ W[K,Nc](f32 -> f16, stored transposed in LDS).
// Block tile 128x128, 8 waves (4x2), each wave 32x64 = 2x4 WMMA tiles.
// LDS double-buffered: one barrier per 32-wide K-step; next tile's global
// loads overlap current tile's WMMAs. A-stream prefetched 3 steps ahead.
// W staged with 4 coalesced b128 loads/thread (2 k-rows x 8 cols), packed
// into b32 transposed LDS stores.
__global__ __launch_bounds__(256)
void wmma_gemm_kernel(const _Float16* __restrict__ A,
                      const float* __restrict__ W,
                      const float* __restrict__ bias,
                      const float* __restrict__ prelu,
                      _Float16* __restrict__ outH,
                      float* __restrict__ outF,
                      int M, int K, int Ncols) {
  __shared__ _Float16 Atile[2][128 * 32];  // 2 x 8 KB
  __shared__ _Float16 Bt[2][128 * 32];     // 2 x 8 KB, transposed: [n][k]

  const int bm = blockIdx.x * 128;
  const int bn = blockIdx.y * 128;
  const int tid = threadIdx.x;
  const int wid = tid >> 5;
  const int lane = tid & 31;
  const int g = lane >> 4;
  const int ln = lane & 15;
  const int wm = wid & 3;   // 4 waves along M (32 rows each)
  const int wn = wid >> 2;  // 2 waves along N (64 cols each)

  // staging roles
  const int arow = tid >> 1;            // A: row 0..127
  const int ahalf = (tid & 1) * 16;     // A: 16-half chunk
  const int wr0 = (tid >> 4) * 2;       // W: k-rows {wr0, wr0+1} (0..30)
  const int wc0 = (tid & 15) * 8;       // W: 8 cols

  v8f acc[2][4] = {};
  const int nk = K >> 5;

  // ---- stage step 0 ----
  {
    *(v16h*)(&Atile[0][arow * 32 + ahalf]) =
        *(const v16h*)(&A[((size_t)(bm + arow)) * K + ahalf]);
    const float* w0 = &W[((size_t)wr0) * Ncols + bn + wc0];
    const float* w1 = &W[((size_t)(wr0 + 1)) * Ncols + bn + wc0];
    v4f a0 = *(const v4f*)(w0),     a1 = *(const v4f*)(w0 + 4);
    v4f b0 = *(const v4f*)(w1),     b1 = *(const v4f*)(w1 + 4);
    union { unsigned u; _Float16 h[2]; } p;
#pragma unroll
    for (int i = 0; i < 4; ++i) {
      p.h[0] = (_Float16)a0[i]; p.h[1] = (_Float16)b0[i];
      *(unsigned*)(&Bt[0][(wc0 + i) * 32 + wr0]) = p.u;
      p.h[0] = (_Float16)a1[i]; p.h[1] = (_Float16)b1[i];
      *(unsigned*)(&Bt[0][(wc0 + 4 + i) * 32 + wr0]) = p.u;
    }
  }
  __syncthreads();

  for (int kk = 0; kk < nk; ++kk) {
    const int cur = kk & 1;
    // ---- stage step kk+1 into the other buffer (overlaps WMMA) ----
    if (kk + 1 < nk) {
      const int nxt = cur ^ 1;
      const int k1 = (kk + 1) * 32;
      *(v16h*)(&Atile[nxt][arow * 32 + ahalf]) =
          *(const v16h*)(&A[((size_t)(bm + arow)) * K + k1 + ahalf]);
      const float* w0 = &W[((size_t)(k1 + wr0)) * Ncols + bn + wc0];
      const float* w1 = &W[((size_t)(k1 + wr0 + 1)) * Ncols + bn + wc0];
      v4f a0 = *(const v4f*)(w0),     a1 = *(const v4f*)(w0 + 4);
      v4f b0 = *(const v4f*)(w1),     b1 = *(const v4f*)(w1 + 4);
      union { unsigned u; _Float16 h[2]; } p;
#pragma unroll
      for (int i = 0; i < 4; ++i) {
        p.h[0] = (_Float16)a0[i]; p.h[1] = (_Float16)b0[i];
        *(unsigned*)(&Bt[nxt][(wc0 + i) * 32 + wr0]) = p.u;
        p.h[0] = (_Float16)a1[i]; p.h[1] = (_Float16)b1[i];
        *(unsigned*)(&Bt[nxt][(wc0 + 4 + i) * 32 + wr0]) = p.u;
      }
    }
    if (kk + 3 < nk)  // gfx1250 global_prefetch_b8 on the A stream
      __builtin_prefetch(&A[((size_t)(bm + arow)) * K + (kk + 3) * 32], 0, 0);

    // ---- fragments from current buffer ----
    union { v16h v; v8h h[2]; } af[2], bf[4];
#pragma unroll
    for (int mt = 0; mt < 2; ++mt) {
      int m = wm * 32 + mt * 16 + ln;
      af[mt].v = *(const v16h*)(&Atile[cur][m * 32 + 16 * g]);       // k=16g..16g+15
    }
#pragma unroll
    for (int nt = 0; nt < 4; ++nt) {
      int n = wn * 64 + nt * 16 + ln;
      bf[nt].v = *(const v16h*)(&Bt[cur][n * 32 + 16 * g]);          // k=16g..16g+15
    }
#pragma unroll
    for (int mt = 0; mt < 2; ++mt)
#pragma unroll
      for (int nt = 0; nt < 4; ++nt)
        acc[mt][nt] = __builtin_amdgcn_wmma_f32_16x16x32_f16(
            false, af[mt].v, false, bf[nt].v, (short)0, acc[mt][nt], false, false);

    __syncthreads();  // single barrier: compute(cur) done + stage(nxt) done
  }

  // ---- epilogue: bias / PReLU / store ----
#pragma unroll
  for (int mt = 0; mt < 2; ++mt) {
#pragma unroll
    for (int nt = 0; nt < 4; ++nt) {
#pragma unroll
      for (int v = 0; v < 8; ++v) {
        int row = bm + wm * 32 + mt * 16 + v + 8 * g;
        int col = bn + wn * 64 + nt * 16 + ln;
        float x = acc[mt][nt][v];
        if (bias) x += bias[col];
        if (prelu) x = fmaxf(x, 0.0f) + prelu[col] * fminf(x, 0.0f);
        size_t idx = (size_t)row * Ncols + col;
        if (outF) outF[idx] = x;
        if (outH) outH[idx] = (_Float16)x;
      }
    }
  }
}

// ------------------------------- attention ---------------------------------
// One block per (16-query block, head, batch). Scores (16x1024 f32) held in
// LDS; masked softmax; P@V via WMMA with V staged transposed (double-
// buffered) in LDS. Output: attnO + x (residual) into ATf.
__global__ __launch_bounds__(256)
void attn_kernel(const _Float16* __restrict__ Qh,
                 const _Float16* __restrict__ Kh,
                 const _Float16* __restrict__ Vh,
                 const int* __restrict__ adj,
                 const float* __restrict__ Xf,
                 float* __restrict__ ATf) {
  extern __shared__ char smem[];
  float*    Sf = (float*)smem;                               // 16*1024*4 = 64 KB
  _Float16* Ph = (_Float16*)(smem + 65536);                  // 16*1024*2 = 32 KB
  _Float16* Qt = (_Float16*)(smem + 65536 + 32768);          // 16*128*2  =  4 KB
  _Float16* Vt = (_Float16*)(smem + 65536 + 32768 + 4096);   // 2x128*32*2= 16 KB

  const int qb = blockIdx.x;
  const int h  = blockIdx.y;
  const int b  = blockIdx.z;
  const int qr0 = qb * 16;
  const int tid = threadIdx.x;
  const int wid = tid >> 5;
  const int lane = tid & 31;
  const int g = lane >> 4;
  const int ln = lane & 15;
  const float scale = 0.044194173824159216f;  // 1/sqrt(512)

  // prefetch the adjacency block rows (16 x 1024 ints) while we do QK^T
  {
    int r = tid >> 4;
    int c = (tid & 15) * 64;
    __builtin_prefetch(&adj[((size_t)b * Nn + (qr0 + r)) * Nn + c], 0, 0);
    __builtin_prefetch(&adj[((size_t)b * Nn + (qr0 + r)) * Nn + c + 32], 0, 0);
  }

  // ---- load Q tile (16 x 128) into LDS ----
  {
    int row = tid >> 4;
    int c0 = (tid & 15) * 8;
    *(v8h*)(&Qt[row * DKn + c0]) =
        *(const v8h*)(&Qh[((size_t)(b * Nn + qr0 + row)) * HSZn + h * DKn + c0]);
  }
  __syncthreads();

  // ---- pass 1: S = Q K^T * scale, masked; each wave covers 128 key cols ----
  {
    v8f sacc[8] = {};
#pragma unroll
    for (int kc = 0; kc < 4; ++kc) {
      int k0 = kc * 32;
      union { v16h v; v8h h[2]; } af;
      af.v = *(const v16h*)(&Qt[ln * DKn + k0 + 16 * g]);
#pragma unroll
      for (int nt = 0; nt < 8; ++nt) {
        int n = wid * 128 + nt * 16 + ln;  // key row index
        v16h bfv = *(const v16h*)(&Kh[((size_t)(b * Nn + n)) * HSZn + h * DKn + k0 + 16 * g]);
        sacc[nt] = __builtin_amdgcn_wmma_f32_16x16x32_f16(
            false, af.v, false, bfv, (short)0, sacc[nt], false, false);
      }
    }
#pragma unroll
    for (int nt = 0; nt < 8; ++nt) {
#pragma unroll
      for (int v = 0; v < 8; ++v) {
        int row = v + 8 * g;
        int col = wid * 128 + nt * 16 + ln;
        float s = sacc[nt][v] * scale;
        int a = adj[((size_t)b * Nn + (qr0 + row)) * Nn + col];
        Sf[row * Nn + col] = a ? s : -1.0e30f;
      }
    }
  }
  __syncthreads();

  // ---- pass 2: row-wise softmax (16 lanes per row) ----
  {
    int r = tid >> 4;
    int j = tid & 15;
    float mx = -1.0e30f;
    for (int c = j; c < Nn; c += 16) mx = fmaxf(mx, Sf[r * Nn + c]);
#pragma unroll
    for (int m = 8; m >= 1; m >>= 1) mx = fmaxf(mx, __shfl_xor(mx, m, 16));
    float sm = 0.0f;
    for (int c = j; c < Nn; c += 16) {
      float e = __expf(Sf[r * Nn + c] - mx);
      sm += e;
      Ph[r * Nn + c] = (_Float16)e;
    }
#pragma unroll
    for (int m = 8; m >= 1; m >>= 1) sm += __shfl_xor(sm, m, 16);
    float inv = 1.0f / sm;
    for (int c = j; c < Nn; c += 16)
      Ph[r * Nn + c] = (_Float16)((float)Ph[r * Nn + c] * inv);
  }

  // ---- pass 3: O = P @ V, double-buffered V chunks, 1 barrier per chunk ----
  // stage: thread handles 2 k-rows x 8 cols; packed b32 stores into Vt[buf]
  const int vr0 = (tid >> 4) * 2;       // k rows {vr0, vr0+1}
  const int vc0 = (tid & 15) * 8;       // 8 dk cols
  {
    union { unsigned u; _Float16 h[2]; } p;
    v8h r0 = *(const v8h*)(&Vh[((size_t)(b * Nn + vr0)) * HSZn + h * DKn + vc0]);
    v8h r1 = *(const v8h*)(&Vh[((size_t)(b * Nn + vr0 + 1)) * HSZn + h * DKn + vc0]);
#pragma unroll
    for (int i = 0; i < 8; ++i) {
      p.h[0] = r0[i]; p.h[1] = r1[i];
      *(unsigned*)(&Vt[(vc0 + i) * 32 + vr0]) = p.u;
    }
  }
  __syncthreads();

  v8f oacc = {};
  for (int kc = 0; kc < 32; ++kc) {
    const int cur = kc & 1;
    if (kc + 1 < 32) {
      const int nxt = cur ^ 1;
      const int k1 = (kc + 1) * 32;
      union { unsigned u; _Float16 h[2]; } p;
      v8h r0 = *(const v8h*)(&Vh[((size_t)(b * Nn + k1 + vr0)) * HSZn + h * DKn + vc0]);
      v8h r1 = *(const v8h*)(&Vh[((size_t)(b * Nn + k1 + vr0 + 1)) * HSZn + h * DKn + vc0]);
#pragma unroll
      for (int i = 0; i < 8; ++i) {
        p.h[0] = r0[i]; p.h[1] = r1[i];
        *(unsigned*)(&Vt[nxt * 4096 + (vc0 + i) * 32 + vr0]) = p.u;
      }
    }
    const int k0 = kc * 32;
    union { v16h v; v8h h[2]; } af;
    af.v = *(const v16h*)(&Ph[ln * Nn + k0 + 16 * g]);
    int c = wid * 16 + ln;
    v16h bfv = *(const v16h*)(&Vt[cur * 4096 + c * 32 + 16 * g]);
    oacc = __builtin_amdgcn_wmma_f32_16x16x32_f16(
        false, af.v, false, bfv, (short)0, oacc, false, false);
    __syncthreads();
  }

  // store with residual (+x)
#pragma unroll
  for (int v = 0; v < 8; ++v) {
    int row = qr0 + v + 8 * g;
    int col = h * DKn + wid * 16 + ln;
    size_t idx = ((size_t)b * Nn + row) * HSZn + col;
    ATf[idx] = oacc[v] + Xf[idx];
  }
}

// ------------------------------- layernorm ---------------------------------
__global__ __launch_bounds__(256)
void ln_kernel(const float* __restrict__ in,
               const float* __restrict__ res,
               const float* __restrict__ sc,
               const float* __restrict__ bi,
               float* __restrict__ outF,
               _Float16* __restrict__ outH) {
  int wid = threadIdx.x >> 5;
  int lane = threadIdx.x & 31;
  size_t row = (size_t)blockIdx.x * 8 + wid;
  const float* src = in + row * HSZn;
  const float* rsp = res ? res + row * HSZn : nullptr;
  float vals[16];
  float s = 0.0f;
#pragma unroll
  for (int i = 0; i < 16; ++i) {
    int c = lane + i * 32;
    float x = src[c];
    if (rsp) x += rsp[c];
    vals[i] = x;
    s += x;
  }
#pragma unroll
  for (int m = 16; m >= 1; m >>= 1) s += __shfl_xor(s, m, 32);
  float mu = s * (1.0f / HSZn);
  float var = 0.0f;
#pragma unroll
  for (int i = 0; i < 16; ++i) {
    float d = vals[i] - mu;
    var += d * d;
  }
#pragma unroll
  for (int m = 16; m >= 1; m >>= 1) var += __shfl_xor(var, m, 32);
  float rinv = rsqrtf(var * (1.0f / HSZn) + 1e-5f);
#pragma unroll
  for (int i = 0; i < 16; ++i) {
    int c = lane + i * 32;
    float y = (vals[i] - mu) * rinv * sc[c] + bi[c];
    outF[row * HSZn + c] = y;
    if (outH) outH[row * HSZn + c] = (_Float16)y;
  }
}

// --------------------------------- output ----------------------------------
__global__ void out_kernel(const float* __restrict__ Xf, float* __restrict__ out) {
  size_t i = (size_t)blockIdx.x * 256 + threadIdx.x;
  const size_t glob_sz = (size_t)Bn * HSZn;                 // 8192
  const size_t node_sz = (size_t)Bn * Nn * HSZn;            // 8388608
  const size_t total = glob_sz + node_sz + (size_t)Bn * Nn; // 8413184
  if (i >= total) return;
  if (i < glob_sz) {
    size_t bb = i >> 9, c = i & 511;
    out[i] = Xf[(bb * Nn + En) * HSZn + c];
  } else if (i < glob_sz + node_sz) {
    out[i] = Xf[i - glob_sz];
  } else {
    out[i] = 1.0f;
  }
}

// --------------------------------- launch ----------------------------------
extern "C" void kernel_launch(void* const* d_in, const int* in_sizes, int n_in,
                              void* d_out, int out_size, void* d_ws, size_t ws_size,
                              hipStream_t stream) {
  const float* ents = (const float*)d_in[0];
  const int*   rels = (const int*)d_in[1];
  const int*   adj  = (const int*)d_in[2];
  const float* renc = (const float*)d_in[3];
  const float* Wq   = (const float*)d_in[4];
  const float* Wk   = (const float*)d_in[5];
  const float* Wv   = (const float*)d_in[6];
  const float* l1w  = (const float*)d_in[7];
  const float* l1b  = (const float*)d_in[8];
  const float* l2w  = (const float*)d_in[9];
  const float* l2b  = (const float*)d_in[10];
  const float* ln1s = (const float*)d_in[11];
  const float* ln1b = (const float*)d_in[12];
  const float* ln2s = (const float*)d_in[13];
  const float* ln2b = (const float*)d_in[14];
  const float* pra  = (const float*)d_in[15];

  char* ws = (char*)d_ws;
  // f32 buffers (each B*N*HSZ*4 = 32 MiB)
  float* Xf  = (float*)(ws + 0);
  float* ATf = (float*)(ws + 33554432);
  float* Tf  = (float*)(ws + 67108864);
  float* Gf  = ATf;  // alias: attention output dead once T is computed
  // f16 buffers (each B*N*HSZ*2 = 16 MiB)
  _Float16* Xh = (_Float16*)(ws + 100663296);
  _Float16* Th = (_Float16*)(ws + 117440512);
  _Float16* Qh = (_Float16*)(ws + 134217728);
  _Float16* Kh = (_Float16*)(ws + 150994944);
  _Float16* Vh = (_Float16*)(ws + 167772160);
  // Hh (B*N*DFF*2 = 64 MiB) aliases Q/K/V region + tail (dead after attention)
  _Float16* Hh = (_Float16*)(ws + 134217728);
  // total footprint: 201326592 bytes

  const int M = Bn * Nn;  // 16384

  gather_kernel<<<Bn * Nn, 256, 0, stream>>>(ents, rels, renc, Xf, Xh);

  for (int j = 0; j < 2; ++j) {
    const float* wq = Wq + (size_t)j * HSZn * HSZn;
    const float* wk = Wk + (size_t)j * HSZn * HSZn;
    const float* wv = Wv + (size_t)j * HSZn * HSZn;

    dim3 g512(M / 128, HSZn / 128);  // (128, 4)
    dim3 gdff(M / 128, DFFn / 128);  // (128, 16)

    wmma_gemm_kernel<<<g512, 256, 0, stream>>>(Xh, wq, nullptr, nullptr, Qh, nullptr, M, HSZn, HSZn);
    wmma_gemm_kernel<<<g512, 256, 0, stream>>>(Xh, wk, nullptr, nullptr, Kh, nullptr, M, HSZn, HSZn);
    wmma_gemm_kernel<<<g512, 256, 0, stream>>>(Xh, wv, nullptr, nullptr, Vh, nullptr, M, HSZn, HSZn);

    dim3 ga(Nn / 16, 4, Bn);  // (64 query blocks, heads, batch)
    attn_kernel<<<ga, 256, 118784, stream>>>(Qh, Kh, Vh, adj, Xf, ATf);

    // t = LN1(attn + x)
    ln_kernel<<<M / 8, 256, 0, stream>>>(ATf, nullptr, ln1s + j * HSZn, ln1b + j * HSZn, Tf, Th);

    // h = PReLU(t @ l1_w + l1_b)   (f16 out)
    wmma_gemm_kernel<<<gdff, 256, 0, stream>>>(Th, l1w + (size_t)j * HSZn * DFFn,
                                               l1b + (size_t)j * DFFn, pra + (size_t)j * DFFn,
                                               Hh, nullptr, M, HSZn, DFFn);
    // g = h @ l2_w + l2_b          (f32 out)
    wmma_gemm_kernel<<<g512, 256, 0, stream>>>(Hh, l2w + (size_t)j * DFFn * HSZn,
                                               l2b + (size_t)j * HSZn, nullptr,
                                               nullptr, Gf, M, DFFn, HSZn);
    // x = LN2(g + t)
    ln_kernel<<<M / 8, 256, 0, stream>>>(Gf, Tf, ln2s + j * HSZn, ln2b + j * HSZn, Xf, Xh);
  }

  const size_t total_out = (size_t)Bn * HSZn + (size_t)Bn * Nn * HSZn + (size_t)Bn * Nn;
  out_kernel<<<(int)((total_out + 255) / 256), 256, 0, stream>>>(Xf, (float*)d_out);
}
